// QloraLinear_21122649162587
// MI455X (gfx1250) — matrix operand order repaired and checked by reading, not compiled, not run on hardware
//
#include <hip/hip_runtime.h>

typedef _Float16 v16h __attribute__((ext_vector_type(16)));
typedef _Float16 v8h  __attribute__((ext_vector_type(8)));
typedef _Float16 v4h  __attribute__((ext_vector_type(4)));
typedef float    v8f  __attribute__((ext_vector_type(8)));
typedef float    v4f  __attribute__((ext_vector_type(4)));

constexpr int kInF  = 4096;
constexpr int kOutF = 4096;
constexpr int kTok  = 8192;   // 4 * 2048
constexpr int kR    = 16;

__device__ __forceinline__ v8f vzero8f() {
  v8f r;
#pragma unroll
  for (int i = 0; i < 8; ++i) r[i] = 0.0f;
  return r;
}

__device__ __forceinline__ v16h cat8(v8h a, v8h b) {
  return __builtin_shufflevector(a, b, 0, 1, 2, 3, 4, 5, 6, 7,
                                       8, 9, 10, 11, 12, 13, 14, 15);
}

// ---------------------------------------------------------------------------
// Prepack: int4-in-int32 weights -> exact f16 (values in [-8,7])
// ---------------------------------------------------------------------------
__global__ __launch_bounds__(256) void qlora_prep_w(
    const int* __restrict__ qw, _Float16* __restrict__ wh) {
  const size_t i = ((size_t)blockIdx.x * 256 + threadIdx.x) * 4;
  const int4 q = *(const int4*)(qw + i);
  v4h h;
  h[0] = (_Float16)(float)q.x;
  h[1] = (_Float16)(float)q.y;
  h[2] = (_Float16)(float)q.z;
  h[3] = (_Float16)(float)q.w;
  *(v4h*)(wh + i) = h;
}

// Prepack: lora_A -> f16, folding LoRA scaling (2.0) in.
__global__ __launch_bounds__(256) void qlora_prep_a(
    const float* __restrict__ a, _Float16* __restrict__ ah) {
  const int i = (blockIdx.x * 256 + threadIdx.x) * 4;
  v4f t = *(const v4f*)(a + i);
  v4h h;
#pragma unroll
  for (int e = 0; e < 4; ++e) h[e] = (_Float16)(2.0f * t[e]);
  *(v4h*)(ah + i) = h;
}

// ---------------------------------------------------------------------------
// xa[m, r] = sum_k x[m,k] * (2*A[r,k]).  One 16x16 WMMA tile per wave.
// ---------------------------------------------------------------------------
__global__ __launch_bounds__(256) void qlora_xa(
    const float* __restrict__ x, const _Float16* __restrict__ ah,
    float* __restrict__ xa) {
  const int tid  = threadIdx.x;
  const int lane = tid & 31;
  const int wave = tid >> 5;
  const int l16  = lane & 15;
  const int kg   = lane >> 4;  // 0 or 1
  const int mrow = blockIdx.x * 128 + wave * 16 + l16;

  v8f c = vzero8f();
  for (int kc = 0; kc < kInF; kc += 32) {
    const float* xp = x + (size_t)mrow * kInF + kc;
    v4f x0 = *(const v4f*)(xp + kg * 8);
    v4f x1 = *(const v4f*)(xp + kg * 8 + 4);
    v4f x2 = *(const v4f*)(xp + 16 + kg * 8);
    v4f x3 = *(const v4f*)(xp + 16 + kg * 8 + 4);
    v16h a;
#pragma unroll
    for (int e = 0; e < 4; ++e) {
      a[e]      = (_Float16)x0[e];
      a[4 + e]  = (_Float16)x1[e];
      a[8 + e]  = (_Float16)x2[e];
      a[12 + e] = (_Float16)x3[e];
    }
    const _Float16* ap = ah + (size_t)l16 * kInF + kc + kg * 8;
    v16h b = cat8(*(const v8h*)(ap), *(const v8h*)(ap + 16));
    c = __builtin_amdgcn_wmma_f32_16x16x32_f16(false, a, false, b, (short)0, c,
                                               false, false);
  }
  const int m0 = blockIdx.x * 128 + wave * 16 + ((lane < 16) ? 0 : 8);
#pragma unroll
  for (int i = 0; i < 8; ++i) xa[(size_t)(m0 + i) * kR + l16] = c[i];
}

// ---------------------------------------------------------------------------
// Main GEMM: out = s[n] * (x @ Q^T) + bias[n] + xa @ B^T
// Block = 128(M) x 128(N), 8 waves. Double-buffered LDS, software-pipelined:
//   - W tile: GLOBAL_LOAD_ASYNC_TO_LDS_B128 (ASYNCcnt), issued 1 chunk ahead
//   - X tile: global->reg (f32) 1 chunk ahead, cvt->f16->LDS after compute
// One barrier per K-chunk.
// ---------------------------------------------------------------------------
__global__ __launch_bounds__(256) void qlora_gemm(
    const float* __restrict__ x, const _Float16* __restrict__ wh,
    const float* __restrict__ qscale, const float* __restrict__ bias,
    const float* __restrict__ xa, const float* __restrict__ loraB,
    float* __restrict__ out) {
  constexpr int KC   = 64;
  constexpr int LROW = 72;          // 64 halves + 8 pad (bank-conflict avoidance)
  constexpr int TSZ  = 128 * LROW;  // halves per buffer
  __shared__ __align__(16) _Float16 sX[2 * TSZ];
  __shared__ __align__(16) _Float16 sW[2 * TSZ];

  const int tid  = threadIdx.x;
  const int lane = tid & 31;
  const int wave = tid >> 5;
  const int l16  = lane & 15;
  const int kg   = lane >> 4;
  const int mblk = blockIdx.y * 128;
  const int nblk = blockIdx.x * 128;

  // staging assignment: 4 threads per row, 16 elements each, rows {srow, srow+64}
  const int srow = tid >> 2;        // 0..63
  const int scol = (tid & 3) * 16;  // column group

  const float*    gx0 = x  + (size_t)(mblk + srow) * kInF + scol;
  const float*    gx1 = x  + (size_t)(mblk + 64 + srow) * kInF + scol;
  const _Float16* gw0 = wh + (size_t)(nblk + srow) * kInF + scol;
  const _Float16* gw1 = wh + (size_t)(nblk + 64 + srow) * kInF + scol;

  v8f acc[8];
#pragma unroll
  for (int t = 0; t < 8; ++t) acc[t] = vzero8f();

  v4f xr[2][4];  // pipelined x staging registers

  auto loadX = [&](int kc) {
#pragma unroll
    for (int j = 0; j < 4; ++j) {
      xr[0][j] = *(const v4f*)(gx0 + kc + j * 4);
      xr[1][j] = *(const v4f*)(gx1 + kc + j * 4);
    }
  };
  auto storeX = [&](int buf) {
#pragma unroll
    for (int p = 0; p < 2; ++p) {
      v8h h0, h1;
#pragma unroll
      for (int j = 0; j < 2; ++j)
#pragma unroll
        for (int e = 0; e < 4; ++e) {
          h0[j * 4 + e] = (_Float16)xr[p][j][e];
          h1[j * 4 + e] = (_Float16)xr[p][j + 2][e];
        }
      _Float16* d = sX + buf * TSZ + (p * 64 + srow) * LROW + scol;
      *(v8h*)(d)     = h0;
      *(v8h*)(d + 8) = h1;
    }
  };
  // Async global->LDS copy of the (already f16) weight tile: 4x b128 per lane.
  auto issueW = [&](int kc, int buf) {
    unsigned l0 = (unsigned)(size_t)(sW + buf * TSZ + srow * LROW + scol);
    unsigned l1 = (unsigned)(size_t)(sW + buf * TSZ + (64 + srow) * LROW + scol);
    unsigned long long g0 = (unsigned long long)(size_t)(gw0 + kc);
    unsigned long long g1 = (unsigned long long)(size_t)(gw1 + kc);
    asm volatile("global_load_async_to_lds_b128 %0, %1, off\n\t"
                 "global_load_async_to_lds_b128 %0, %1, off offset:16"
                 :: "v"(l0), "v"(g0) : "memory");
    asm volatile("global_load_async_to_lds_b128 %0, %1, off\n\t"
                 "global_load_async_to_lds_b128 %0, %1, off offset:16"
                 :: "v"(l1), "v"(g1) : "memory");
  };
  auto compute = [&](int buf) {
#pragma unroll
    for (int s = 0; s < 2; ++s) {
      const int ks = s * 32;
      const _Float16* ax =
          sX + buf * TSZ + (wave * 16 + l16) * LROW + ks + kg * 8;
      const v16h afrag = cat8(*(const v8h*)(ax), *(const v8h*)(ax + 16));
#pragma unroll
      for (int nt = 0; nt < 8; ++nt) {
        const _Float16* bx =
            sW + buf * TSZ + (nt * 16 + l16) * LROW + ks + kg * 8;
        const v16h bfrag = cat8(*(const v8h*)(bx), *(const v8h*)(bx + 16));
        acc[nt] = __builtin_amdgcn_wmma_f32_16x16x32_f16(
            false, afrag, false, bfrag, (short)0, acc[nt], false, false);
      }
    }
  };

  // ---- prologue: stage chunk 0 into buffer 0 ----
  issueW(0, 0);
  loadX(0);
  storeX(0);
  asm volatile("s_wait_asynccnt 0x0" ::: "memory");
  __syncthreads();

  constexpr int NCHUNK = kInF / KC;  // 64
  for (int i = 0; i < NCHUNK; ++i) {
    const int buf = i & 1;
    const int nxt = buf ^ 1;
    const int kc  = i * KC;
    if (i < NCHUNK - 1) {
      issueW(kc + KC, nxt);   // async W -> LDS[nxt], lands during compute
      loadX(kc + KC);         // x -> regs, lands during compute
      if (i < NCHUNK - 2) {   // deep prefetch of the x stream (global_prefetch_b8)
        __builtin_prefetch(gx0 + kc + 2 * KC, 0, 3);
        __builtin_prefetch(gx1 + kc + 2 * KC, 0, 3);
      }
    }
    compute(buf);
    if (i < NCHUNK - 1) {
      storeX(nxt);            // cvt + ds_store into the free buffer
      asm volatile("s_wait_asynccnt 0x0" ::: "memory");
    }
    __syncthreads();          // chunk i reads done by all; nxt buffer complete
  }

  // -------- epilogue: scale by qscale, LoRA rank-16 WMMA, bias, store --------
  v16h af;
#pragma unroll
  for (int i = 0; i < 16; ++i) af[i] = (_Float16)0.0f;
  {
    const float* xp = xa + (size_t)(mblk + wave * 16 + l16) * kR + kg * 8;
    v4f a0 = *(const v4f*)(xp);
    v4f a1 = *(const v4f*)(xp + 4);
#pragma unroll
    for (int e = 0; e < 4; ++e) {
      af[e]     = (_Float16)a0[e];
      af[4 + e] = (_Float16)a1[e];
    }
  }

#pragma unroll
  for (int nt = 0; nt < 8; ++nt) {
    const int n    = nblk + nt * 16 + l16;
    const float sv = qscale[n];
    const float bv = bias[n];

    v16h bf;
#pragma unroll
    for (int i = 0; i < 16; ++i) bf[i] = (_Float16)0.0f;
    {  // lane half kg covers K=8*kg..8*kg+7 in elements 0..7; K=16..31 zero-padded
      const v4f* bp = (const v4f*)(loraB + (size_t)n * kR + kg * 8);
#pragma unroll
      for (int q = 0; q < 2; ++q) {
        v4f t = bp[q];
#pragma unroll
        for (int e = 0; e < 4; ++e) bf[q * 4 + e] = (_Float16)t[e];
      }
    }

    v8f c = acc[nt];
#pragma unroll
    for (int i = 0; i < 8; ++i) c[i] *= sv;
    c = __builtin_amdgcn_wmma_f32_16x16x32_f16(false, af, false, bf, (short)0,
                                               c, false, false);
#pragma unroll
    for (int i = 0; i < 8; ++i) c[i] += bv;

    const int mr = mblk + wave * 16 + ((lane < 16) ? 0 : 8);
    float* op = out + (size_t)mr * kOutF + n;
#pragma unroll
    for (int i = 0; i < 8; ++i) op[(size_t)i * kOutF] = c[i];
  }
}

// ---------------------------------------------------------------------------
extern "C" void kernel_launch(void* const* d_in, const int* in_sizes, int n_in,
                              void* d_out, int out_size, void* d_ws,
                              size_t ws_size, hipStream_t stream) {
  (void)in_sizes; (void)n_in; (void)out_size; (void)ws_size;
  const float* x      = (const float*)d_in[0];
  const int*   qw     = (const int*)d_in[1];
  const float* qscale = (const float*)d_in[2];
  const float* bias   = (const float*)d_in[3];
  const float* loraA  = (const float*)d_in[4];
  const float* loraB  = (const float*)d_in[5];
  float* out = (float*)d_out;

  char* ws = (char*)d_ws;
  _Float16* wh = (_Float16*)ws;                                   // 32 MB
  _Float16* ah = (_Float16*)(ws + (size_t)kOutF * kInF * 2);      // 128 KB
  float*    xa = (float*)(ws + (size_t)kOutF * kInF * 2
                             + (size_t)kR * kInF * 2);            // 512 KB

  qlora_prep_w<<<(kOutF * kInF) / (256 * 4), 256, 0, stream>>>(qw, wh);
  qlora_prep_a<<<(kR * kInF) / (256 * 4), 256, 0, stream>>>(loraA, ah);
  qlora_xa<<<kTok / 128, 256, 0, stream>>>(x, ah, xa);
  qlora_gemm<<<dim3(kOutF / 128, kTok / 128), 256, 0, stream>>>(
      x, wh, qscale, bias, xa, loraB, out);
}
